// ingr_embed_59133109731341
// MI455X (gfx1250) — matrix operand order repaired and checked by reading, not compiled, not run on hardware
//
#include <hip/hip_runtime.h>
#include <hip/hip_bf16.h>

typedef __bf16 bf16;
typedef __attribute__((ext_vector_type(16))) __bf16 v16bf;
typedef __attribute__((ext_vector_type(8)))  __bf16 v8bf;
typedef __attribute__((ext_vector_type(8)))  float  v8f;
typedef __attribute__((ext_vector_type(4)))  unsigned int u32x4;
typedef __attribute__((ext_vector_type(8)))  int i32x8;
typedef __attribute__((ext_vector_type(4)))  int i32x4;

#define TOKENS  81920      // 4096 * 20
#define LSEQ    20
#define BATCH   4096
#define H1      512
#define H2      1024
#define NOUT    2047
#define NPAD    2048

// ---------------------------------------------------------------------------
// Kernel 1: convert + transpose weights to bf16, N-major (B^T), pad W3 to 2048
// ---------------------------------------------------------------------------
__global__ void __launch_bounds__(256)
prep_w_kernel(const float* __restrict__ W2, const float* __restrict__ W3,
              bf16* __restrict__ W2bT, bf16* __restrict__ W3bT) {
    int i = blockIdx.x * blockDim.x + threadIdx.x;
    if (i < H2 * H1) {                       // W2bT[n][k] = W2[k][n]
        int n = i / H1, k = i % H1;
        W2bT[i] = (bf16)W2[(size_t)k * H2 + n];
    }
    if (i < NPAD * H2) {                     // W3bT[n][k] = W3[k][n], n==2047 -> 0
        int n = i / H2, k = i % H2;
        float v = (n < NOUT) ? W3[(size_t)k * NOUT + n] : 0.0f;
        W3bT[i] = (bf16)v;
    }
}

// ---------------------------------------------------------------------------
// Kernel 2: h1[token][k] = bf16(relu(W1[ingr[token]][k] + b1[k]))
// ---------------------------------------------------------------------------
__global__ void __launch_bounds__(256)
gather_h1_kernel(const int* __restrict__ ingrs, const float* __restrict__ W1,
                 const float* __restrict__ b1, bf16* __restrict__ h1b) {
    int token = blockIdx.x;
    int idx = ingrs[token];
    const float* src = W1 + (size_t)idx * H1;
    bf16* dst = h1b + (size_t)token * H1;
    for (int c = threadIdx.x; c < H1; c += blockDim.x) {
        float v = src[c] + b1[c];
        v = v > 0.0f ? v : 0.0f;
        dst[c] = (bf16)v;
    }
}

// ---------------------------------------------------------------------------
// TDM: async 2D tile (128 rows x 64B) global -> LDS, dword elements,
// LDS row padded by 4 dwords every 16 dwords (80B stride, conflict-free).
// Descriptor per CDNA5 ISA ch.8 (D# groups 0/1; groups 2/3 zero for 2D).
// ---------------------------------------------------------------------------
__device__ __forceinline__ void tdm_load_tile(unsigned lds_off, const bf16* gptr,
                                              int K, unsigned tensorRows) {
    unsigned long long ga = (unsigned long long)(size_t)gptr;
    unsigned strideDw = (unsigned)(K >> 1);       // row stride & dim0 length (dwords)
    u32x4 g0;
    g0[0] = 1u;                                    // count=1 (valid), user desc
    g0[1] = lds_off;                               // lds_addr (bytes)
    g0[2] = (unsigned)(ga & 0xffffffffu);          // global_addr[31:0]
    g0[3] = (unsigned)((ga >> 32) & 0x01ffffffu)   // global_addr[56:32]
          | (2u << 30);                            // type = 2 ("image")
    i32x8 g1;
    g1[0] = (int)((2u << 16)       // data_size = 2 -> 4-byte elements
                | (1u << 20)       // pad_enable
                | (3u << 22)       // pad_interval: every 16 dwords
                | (3u << 25));     // pad_amount: 4 dwords (16B)
    g1[1] = (int)((strideDw & 0xffffu) << 16);                 // tensor_dim0 lo16
    g1[2] = (int)((strideDw >> 16) | ((tensorRows & 0xffffu) << 16)); // dim0 hi | dim1 lo
    g1[3] = (int)((tensorRows >> 16) | (16u << 16));           // dim1 hi | tile_dim0=16 dw
    g1[4] = 128;                                               // tile_dim1=128, tile_dim2=0
    g1[5] = (int)strideDw;                                     // tensor_dim0_stride lo32
    g1[6] = 0;                                                 // stride hi | dim1_stride lo
    g1[7] = 0;
    i32x4 gz4 = {0, 0, 0, 0};
    i32x8 gz8 = {0, 0, 0, 0, 0, 0, 0, 0};
    __builtin_amdgcn_tensor_load_to_lds(g0, g1, gz4, gz4, gz8, 0);
}

// ---------------------------------------------------------------------------
// Tiled bf16 GEMM: C[M][N](bf16) = A[M][K] * Bt[N][K]^T + bias, optional ReLU.
// Block tile 128x128, 256 threads = 8 waves in 4(M)x2(N) grid.
// TDM double-buffered LDS staging; 8 v_wmma_f32_16x16x32_bf16 per K step.
// ---------------------------------------------------------------------------
template <bool RELU>
__global__ void __launch_bounds__(256)
gemm_bf16_kernel(const bf16* __restrict__ A,   // [M][K], lda = K
                 const bf16* __restrict__ Bt,  // [N][K], ldb = K  (B transposed)
                 const float* __restrict__ bias, int biasN,
                 bf16* __restrict__ C, int ldc, int K, int Ntot) {
    __shared__ __align__(16) bf16 lA[2][128][40];   // 32 K + 8 pad (80B stride)
    __shared__ __align__(16) bf16 lB[2][128][40];

    const int tid   = threadIdx.x;
    const int lane  = tid & 31;
    const int wave  = tid >> 5;
    const int waveM = wave & 3;          // 0..3 -> 32-row strip
    const int waveN = wave >> 2;         // 0..1 -> 64-col strip
    const int lm    = lane & 15;
    const int lh    = lane >> 4;
    const int m0    = blockIdx.y * 128;
    const int n0    = blockIdx.x * 128;

    const unsigned ldsA0 = (unsigned)(size_t)&lA[0][0][0];
    const unsigned ldsA1 = (unsigned)(size_t)&lA[1][0][0];
    const unsigned ldsB0 = (unsigned)(size_t)&lB[0][0][0];
    const unsigned ldsB1 = (unsigned)(size_t)&lB[1][0][0];

    v8f acc[2][4];
#pragma unroll
    for (int i = 0; i < 2; ++i)
#pragma unroll
        for (int j = 0; j < 4; ++j) acc[i][j] = {};

    union Frag { v16bf v; v8bf h[2]; };

    // prologue: TDM-load first tiles into buffer 0 (wave 0 issues; TDM is per-wave)
    if (wave == 0) {
        tdm_load_tile(ldsA0, A + (size_t)m0 * K, K, (unsigned)TOKENS);
        tdm_load_tile(ldsB0, Bt + (size_t)n0 * K, K, (unsigned)Ntot);
        __builtin_amdgcn_s_wait_tensorcnt(0);
    }
    __syncthreads();

    int cur = 0;
    for (int k0 = 0; k0 < K; k0 += 32) {
        // async-prefetch next K tile into the other buffer
        if (k0 + 32 < K && wave == 0) {
            unsigned la = cur ? ldsA0 : ldsA1;
            unsigned lb = cur ? ldsB0 : ldsB1;
            tdm_load_tile(la, A + (size_t)m0 * K + (k0 + 32), K, (unsigned)TOKENS);
            tdm_load_tile(lb, Bt + (size_t)n0 * K + (k0 + 32), K, (unsigned)Ntot);
        }

        Frag a[2], b[4];
#pragma unroll
        for (int mt = 0; mt < 2; ++mt) {
            int r = waveM * 32 + mt * 16 + lm;     // A row (M), per ISA A layout
            a[mt].h[0] = *(const v8bf*)&lA[cur][r][8 * lh];
            a[mt].h[1] = *(const v8bf*)&lA[cur][r][16 + 8 * lh];
        }
#pragma unroll
        for (int nt = 0; nt < 4; ++nt) {
            int c = waveN * 64 + nt * 16 + lm;     // B col (N), same K order as A
            b[nt].h[0] = *(const v8bf*)&lB[cur][c][8 * lh];
            b[nt].h[1] = *(const v8bf*)&lB[cur][c][16 + 8 * lh];
        }
#pragma unroll
        for (int mt = 0; mt < 2; ++mt)
#pragma unroll
            for (int nt = 0; nt < 4; ++nt)
                acc[mt][nt] = __builtin_amdgcn_wmma_f32_16x16x32_bf16(
                    false, a[mt].v, false, b[nt].v,
                    (short)0, acc[mt][nt], false, false);

        // one barrier per step: next buffer loaded (wave0 tensorcnt) AND
        // current buffer free for overwrite (all waves' ds reads done)
        if (wave == 0) __builtin_amdgcn_s_wait_tensorcnt(0);
        __syncthreads();
        cur ^= 1;
    }

    // epilogue: C layout -> row = r + 8*lh, col = lane%16
#pragma unroll
    for (int mt = 0; mt < 2; ++mt) {
#pragma unroll
        for (int nt = 0; nt < 4; ++nt) {
            int col = n0 + waveN * 64 + nt * 16 + lm;
            float bs = (col < biasN) ? bias[col] : 0.0f;
#pragma unroll
            for (int r = 0; r < 8; ++r) {
                int row = m0 + waveM * 32 + mt * 16 + r + 8 * lh;
                float v = acc[mt][nt][r] + bs;
                if (RELU) v = v > 0.0f ? v : 0.0f;
                C[(size_t)row * ldc + col] = (bf16)v;
            }
        }
    }
}

// ---------------------------------------------------------------------------
// Kernel 5: per-token inverse L2 norm of emb rows
// ---------------------------------------------------------------------------
__global__ void __launch_bounds__(256)
inv_norm_kernel(const bf16* __restrict__ embb, float* __restrict__ inv) {
    int token = blockIdx.x;
    const bf16* row = embb + (size_t)token * NPAD;
    float ss = 0.0f;
    for (int c = threadIdx.x; c < NOUT; c += 256) {
        float v = (float)row[c];
        ss += v * v;
    }
    __shared__ float red[256];
    red[threadIdx.x] = ss;
    __syncthreads();
    for (int s = 128; s > 0; s >>= 1) {
        if (threadIdx.x < s) red[threadIdx.x] += red[threadIdx.x + s];
        __syncthreads();
    }
    if (threadIdx.x == 0) {
        float n = sqrtf(red[0]);
        inv[token] = 1.0f / fmaxf(n, 1e-12f);
    }
}

// ---------------------------------------------------------------------------
// Kernel 6: out[b][o] = sum_{l < len[b]} emb[b*20+l][o] * inv[b*20+l]
// ---------------------------------------------------------------------------
__global__ void __launch_bounds__(256)
reduce_kernel(const bf16* __restrict__ embb, const float* __restrict__ inv,
              const int* __restrict__ lengths, float* __restrict__ out) {
    int b = blockIdx.x;
    __shared__ float w[LSEQ];
    if (threadIdx.x < LSEQ) {
        int len = lengths[b];
        w[threadIdx.x] = (threadIdx.x < len) ? inv[b * LSEQ + threadIdx.x] : 0.0f;
    }
    __syncthreads();
    const bf16* base = embb + (size_t)b * LSEQ * NPAD;
    for (int o = threadIdx.x; o < NOUT; o += 256) {
        float acc = 0.0f;
#pragma unroll
        for (int l = 0; l < LSEQ; ++l)
            acc += (float)base[(size_t)l * NPAD + o] * w[l];
        out[(size_t)b * NOUT + o] = acc;
    }
}

// ---------------------------------------------------------------------------
extern "C" void kernel_launch(void* const* d_in, const int* in_sizes, int n_in,
                              void* d_out, int out_size, void* d_ws, size_t ws_size,
                              hipStream_t stream) {
    const int*   ingrs   = (const int*)d_in[0];
    const int*   lengths = (const int*)d_in[1];
    const float* W1      = (const float*)d_in[2];
    const float* b1      = (const float*)d_in[3];
    const float* W2      = (const float*)d_in[4];
    const float* b2      = (const float*)d_in[5];
    const float* W3      = (const float*)d_in[6];
    const float* b3      = (const float*)d_in[7];
    float* out = (float*)d_out;

    char* ws = (char*)d_ws;
    size_t off = 0;
    bf16* W2bT = (bf16*)(ws + off); off += (size_t)H2 * H1 * 2;        // 1 MB
    bf16* W3bT = (bf16*)(ws + off); off += (size_t)NPAD * H2 * 2;      // 4 MB
    bf16* h1b  = (bf16*)(ws + off); off += (size_t)TOKENS * H1 * 2;    // 84 MB
    bf16* h2b  = (bf16*)(ws + off); off += (size_t)TOKENS * H2 * 2;    // 168 MB
    bf16* embb = (bf16*)(ws + off); off += (size_t)TOKENS * NPAD * 2;  // 336 MB
    float* inv = (float*)(ws + off);

    // 1. weight conversion/transposition
    prep_w_kernel<<<(NPAD * H2 + 255) / 256, 256, 0, stream>>>(W2, W3, W2bT, W3bT);
    // 2. embedding gather + bias + relu -> bf16
    gather_h1_kernel<<<TOKENS, 256, 0, stream>>>(ingrs, W1, b1, h1b);
    // 3. GEMM1: h2 = relu(h1 @ W2 + b2)
    dim3 g1(H2 / 128, TOKENS / 128);
    gemm_bf16_kernel<true><<<g1, 256, 0, stream>>>(h1b, W2bT, b2, H2, h2b, H2, H1, H2);
    // 4. GEMM2: emb = h2 @ W3 + b3 (N padded to 2048; col 2047 = 0)
    dim3 g2(NPAD / 128, TOKENS / 128);
    gemm_bf16_kernel<false><<<g2, 256, 0, stream>>>(h2b, W3bT, b3, NOUT, embb, NPAD, H2, NPAD);
    // 5. per-token 1/||emb||
    inv_norm_kernel<<<TOKENS, 256, 0, stream>>>(embb, inv);
    // 6. masked normalized sum over L
    reduce_kernel<<<BATCH, 256, 0, stream>>>(embb, inv, lengths, out);
}
